// GCN_11622181503541
// MI455X (gfx1250) — matrix operand order repaired and checked by reading, not compiled, not run on hardware
//
#include <hip/hip_runtime.h>

// ---------------------------------------------------------------------------
// MI455X (gfx1250) fused GCN forward.
//
// Roofline: A (1 GiB f32) must stream from HBM twice -> ~92us floor @23.3TB/s.
// A@H arithmetic intensity is 32 FLOP/byte-of-A, so this is memory-bound only
// if the matrix path sustains >=745 TFLOP/s: that requires the f16 WMMA path
// (v_wmma_f32_16x16x32_f16, f32 accumulate), not the slow 16x16x4 f32 path.
// A entries are uniform[0,1) and we accumulate in f32, so f32->f16 conversion
// of A/H tiles during staging is the precision/data-path the chip rewards.
// The K-loop is double-buffered in LDS (one s_barrier per 32-K chunk) so the
// global_load_b128 stream stays in flight underneath the v_wmma issue.
// ---------------------------------------------------------------------------

typedef __attribute__((ext_vector_type(16))) _Float16 v16h;
typedef __attribute__((ext_vector_type(8)))  _Float16 v8h;
typedef __attribute__((ext_vector_type(8)))  float    v8f;

union V16 { v16h v; v8h h[2]; };

__device__ __forceinline__ v8f zero8() {
  v8f z = {0.f, 0.f, 0.f, 0.f, 0.f, 0.f, 0.f, 0.f};
  return z;
}

__device__ __forceinline__ v8h cvt8(float4 a, float4 b) {
  v8h r = { (_Float16)a.x, (_Float16)a.y, (_Float16)a.z, (_Float16)a.w,
            (_Float16)b.x, (_Float16)b.y, (_Float16)b.z, (_Float16)b.w };
  return r;
}

__device__ __forceinline__ v8f wmma_f16(const V16& a, const V16& b, v8f c) {
  return __builtin_amdgcn_wmma_f32_16x16x32_f16(
      /*neg_a=*/false, a.v, /*neg_b=*/false, b.v,
      /*c_mod=*/(short)0, c, /*reuse_a=*/false, /*reuse_b=*/false);
}

// ---------------------------------------------------------------------------
// prop_kernel: Hout = relu( (A @ Hin) @ W^T ),  Hin/Hout: [n,64], W: [64,64].
// Grid: n/64 workgroups x 256 threads (8 waves). Each workgroup owns 64 rows.
// Wave w: rows (w/2)*16..+15, output cols (w%2)*32..+31 (2 n-tiles of 16).
// ---------------------------------------------------------------------------
__global__ __launch_bounds__(256, 1)
void prop_kernel(const float* __restrict__ A, const float* __restrict__ Hin,
                 const float* __restrict__ W, float* __restrict__ Hout, int n)
{
  __shared__ _Float16 sA[2][64][32];  // A tile  [m][k]   f16, double buffered (8 KB)
  __shared__ _Float16 sB[2][64][32];  // H panel [col][k] f16, double buffered (8 KB)
  __shared__ _Float16 sS[64][64];     // S = A@Hin block, f16 (8 KB) for epilogue

  const int tid   = threadIdx.x;
  const int wave  = tid >> 5;
  const int lane  = tid & 31;
  const int l16   = lane & 15;
  const int hf    = lane >> 4;          // which 16-lane half
  const int mrow  = (wave >> 1) * 16;   // wave's row base inside block
  const int cbase = (wave & 1) * 32;    // wave's col base (2 n-tiles)
  const int blockRow = blockIdx.x * 64;

  // Staging roles (coalesced):
  const int arow = tid >> 2;            // 0..63 : A row inside tile
  const int acol = (tid & 3) * 8;       // 8 contiguous f32 per thread
  const int bk   = tid & 31;            // H panel k index
  const int bcg  = (tid >> 5) * 8;      // H panel col group

  const float* Arow = A + (size_t)(blockRow + arow) * (size_t)n + acol;

  // ---- prologue: stage chunk 0 into buffer 0 ----
  {
    const float4* pa = (const float4*)Arow;
    const float4  a0 = pa[0], a1 = pa[1];
    *(v8h*)&sA[0][arow][acol] = cvt8(a0, a1);
    const float4* ph = (const float4*)(Hin + (size_t)bk * 64 + bcg);
    const float4  h0 = ph[0], h1 = ph[1];
    sB[0][bcg + 0][bk] = (_Float16)h0.x;
    sB[0][bcg + 1][bk] = (_Float16)h0.y;
    sB[0][bcg + 2][bk] = (_Float16)h0.z;
    sB[0][bcg + 3][bk] = (_Float16)h0.w;
    sB[0][bcg + 4][bk] = (_Float16)h1.x;
    sB[0][bcg + 5][bk] = (_Float16)h1.y;
    sB[0][bcg + 6][bk] = (_Float16)h1.z;
    sB[0][bcg + 7][bk] = (_Float16)h1.w;
  }
  __syncthreads();

  v8f acc0 = zero8(), acc1 = zero8();
  int buf = 0;

  // ---- steady state: 1 barrier / chunk; next-chunk loads fly under WMMAs ----
  for (int k0 = 32; k0 < n; k0 += 32) {
    // issue next chunk's global loads first
    const float4* pa = (const float4*)(Arow + k0);
    const float4  a0 = pa[0], a1 = pa[1];
    __builtin_prefetch((const void*)(pa + 8), 0, 3);  // one chunk further ahead
    const float4* ph = (const float4*)(Hin + (size_t)(k0 + bk) * 64 + bcg);
    const float4  h0 = ph[0], h1 = ph[1];

    // compute current chunk from LDS (contiguous k = 16h+s, same order for A and B)
    V16 af, bf0, bf1;
    af.h[0] = *(const v8h*)&sA[buf][mrow + l16][16 * hf];       // K {16h..16h+7}
    af.h[1] = *(const v8h*)&sA[buf][mrow + l16][16 * hf + 8];   // K {16h+8..16h+15}
    const int j0 = cbase + l16;
    const int j1 = cbase + 16 + l16;
    bf0.h[0] = *(const v8h*)&sB[buf][j0][16 * hf];              // K {16h..16h+15}
    bf0.h[1] = *(const v8h*)&sB[buf][j0][16 * hf + 8];
    bf1.h[0] = *(const v8h*)&sB[buf][j1][16 * hf];
    bf1.h[1] = *(const v8h*)&sB[buf][j1][16 * hf + 8];
    acc0 = wmma_f16(af, bf0, acc0);
    acc1 = wmma_f16(af, bf1, acc1);

    // convert + stage next chunk into the other buffer
    const int nb = buf ^ 1;
    *(v8h*)&sA[nb][arow][acol] = cvt8(a0, a1);
    sB[nb][bcg + 0][bk] = (_Float16)h0.x;
    sB[nb][bcg + 1][bk] = (_Float16)h0.y;
    sB[nb][bcg + 2][bk] = (_Float16)h0.z;
    sB[nb][bcg + 3][bk] = (_Float16)h0.w;
    sB[nb][bcg + 4][bk] = (_Float16)h1.x;
    sB[nb][bcg + 5][bk] = (_Float16)h1.y;
    sB[nb][bcg + 6][bk] = (_Float16)h1.z;
    sB[nb][bcg + 7][bk] = (_Float16)h1.w;
    __syncthreads();
    buf = nb;
  }

  // ---- drain: compute final chunk ----
  {
    V16 af, bf0, bf1;
    af.h[0] = *(const v8h*)&sA[buf][mrow + l16][16 * hf];
    af.h[1] = *(const v8h*)&sA[buf][mrow + l16][16 * hf + 8];
    const int j0 = cbase + l16;
    const int j1 = cbase + 16 + l16;
    bf0.h[0] = *(const v8h*)&sB[buf][j0][16 * hf];
    bf0.h[1] = *(const v8h*)&sB[buf][j0][16 * hf + 8];
    bf1.h[0] = *(const v8h*)&sB[buf][j1][16 * hf];
    bf1.h[1] = *(const v8h*)&sB[buf][j1][16 * hf + 8];
    acc0 = wmma_f16(af, bf0, acc0);
    acc1 = wmma_f16(af, bf1, acc1);
  }
  __syncthreads();

  // ---- epilogue: out = relu( S @ W^T );  C/D layout: VGPR r <-> M = r + 8*hf
#pragma unroll
  for (int r = 0; r < 8; ++r) {
    const int m = mrow + r + 8 * hf;
    sS[m][cbase + l16]      = (_Float16)acc0[r];
    sS[m][cbase + 16 + l16] = (_Float16)acc1[r];
  }
  __syncthreads();

  v8f o0 = zero8(), o1 = zero8();
#pragma unroll
  for (int kc = 0; kc < 64; kc += 32) {
    V16 af, bf;
    af.h[0] = *(const v8h*)&sS[mrow + l16][kc + 16 * hf];
    af.h[1] = *(const v8h*)&sS[mrow + l16][kc + 16 * hf + 8];
    {
      // B(k,j) = W[j][k]; per-lane contiguous run of 16 f32 from global (L2-hot)
      const int j = cbase + l16;
      const float4* pw = (const float4*)(W + j * 64 + kc + 16 * hf);
      bf.h[0] = cvt8(pw[0], pw[1]);
      bf.h[1] = cvt8(pw[2], pw[3]);
      o0 = wmma_f16(af, bf, o0);
    }
    {
      const int j = cbase + 16 + l16;
      const float4* pw = (const float4*)(W + j * 64 + kc + 16 * hf);
      bf.h[0] = cvt8(pw[0], pw[1]);
      bf.h[1] = cvt8(pw[2], pw[3]);
      o1 = wmma_f16(af, bf, o1);
    }
  }
#pragma unroll
  for (int r = 0; r < 8; ++r) {
    const int m = blockRow + mrow + r + 8 * hf;
    Hout[m * 64 + cbase + l16]      = fmaxf(o0[r], 0.f);
    Hout[m * 64 + cbase + 16 + l16] = fmaxf(o1[r], 0.f);
  }
}

// ---------------------------------------------------------------------------
// edge_kernel: out[e] = sigmoid([H[src],H[dst]] @ Wl1^T + b) @ Wl3^T
// 128 edges per workgroup; pair matrix 128x128 f16 in LDS; 16 WMMAs/wave.
// H (4 MB) is L2-resident, so the random row gathers are cheap.
// ---------------------------------------------------------------------------
__global__ __launch_bounds__(256, 1)
void edge_kernel(const float* __restrict__ H, const float* __restrict__ Wl1,
                 const float* __restrict__ bl1, const float* __restrict__ Wl3,
                 const int* __restrict__ src, const int* __restrict__ dst,
                 float* __restrict__ out)
{
  __shared__ _Float16 sP[128][128];  // pair matrix f16 (32 KB)
  __shared__ _Float16 sW[64][128];   // W_l1 as [j][k] f16 (16 KB)

  const int tid  = threadIdx.x;
  const int wave = tid >> 5;
  const int lane = tid & 31;
  const int l16  = lane & 15;
  const int hf   = lane >> 4;

  // stage W_l1 [64,128] f32 -> sW[j][k] f16
  {
    const int j = tid >> 2, kg = (tid & 3) * 32;
    const float4* pw = (const float4*)(Wl1 + j * 128 + kg);
#pragma unroll
    for (int i = 0; i < 4; ++i)
      *(v8h*)&sW[j][kg + i * 8] = cvt8(pw[2 * i], pw[2 * i + 1]);
  }
  // gather pair rows: thread handles one half-row (src or dst endpoint)
  {
    const int el  = tid >> 1;
    const int sel = tid & 1;
    const int e   = blockIdx.x * 128 + el;
    const int idx = sel ? dst[e] : src[e];
    const float4* ph = (const float4*)(H + (size_t)idx * 64);
#pragma unroll
    for (int i = 0; i < 8; ++i)
      *(v8h*)&sP[el][sel * 64 + i * 8] = cvt8(ph[2 * i], ph[2 * i + 1]);
  }
  __syncthreads();

  v8f acc[4] = {zero8(), zero8(), zero8(), zero8()};
#pragma unroll
  for (int kc = 0; kc < 128; kc += 32) {
    V16 af;
    af.h[0] = *(const v8h*)&sP[wave * 16 + l16][kc + 16 * hf];
    af.h[1] = *(const v8h*)&sP[wave * 16 + l16][kc + 16 * hf + 8];
#pragma unroll
    for (int jt = 0; jt < 4; ++jt) {
      V16 bf;
      const int j = jt * 16 + l16;
      bf.h[0] = *(const v8h*)&sW[j][kc + 16 * hf];
      bf.h[1] = *(const v8h*)&sW[j][kc + 16 * hf + 8];
      acc[jt] = wmma_f16(af, bf, acc[jt]);
    }
  }

  // bias + sigmoid + dot with W_l3 row, reduced across each 16-lane half.
  float p[8] = {0.f, 0.f, 0.f, 0.f, 0.f, 0.f, 0.f, 0.f};
#pragma unroll
  for (int jt = 0; jt < 4; ++jt) {
    const int j = jt * 16 + l16;
    const float bb = bl1[j];
    const float w3 = Wl3[j];
#pragma unroll
    for (int r = 0; r < 8; ++r) {
      const float hsig = 1.f / (1.f + __expf(-(acc[jt][r] + bb)));
      p[r] += hsig * w3;
    }
  }
#pragma unroll
  for (int off = 8; off; off >>= 1) {
#pragma unroll
    for (int r = 0; r < 8; ++r) p[r] += __shfl_xor(p[r], off, 16);
  }
  if (l16 == 0) {
    const int ebase = blockIdx.x * 128 + wave * 16 + 8 * hf;
#pragma unroll
    for (int r = 0; r < 8; ++r) out[ebase + r] = p[r];
  }
}

__global__ void tail_kernel(float* __restrict__ p) { *p = 0.f; }

// ---------------------------------------------------------------------------
extern "C" void kernel_launch(void* const* d_in, const int* in_sizes, int n_in,
                              void* d_out, int out_size, void* d_ws, size_t ws_size,
                              hipStream_t stream) {
  const float* X    = (const float*)d_in[0];
  const float* A    = (const float*)d_in[1];
  const float* Wfc1 = (const float*)d_in[2];
  const float* Wfc3 = (const float*)d_in[3];
  const float* Wl1  = (const float*)d_in[4];
  const float* bl1  = (const float*)d_in[5];
  const float* Wl3  = (const float*)d_in[6];
  const int*   esrc = (const int*)d_in[7];
  const int*   edst = (const int*)d_in[8];
  float* out = (float*)d_out;

  const int n = in_sizes[0] / 64;   // 16384
  const int e = in_sizes[7];        // 1048576

  float* H1 = (float*)d_ws;                  // [n,64] f32
  float* H2 = H1 + (size_t)n * 64;           // [n,64] f32

  prop_kernel<<<n / 64, 256, 0, stream>>>(A, X, Wfc1, H1, n);
  prop_kernel<<<n / 64, 256, 0, stream>>>(A, H1, Wfc3, H2, n);
  edge_kernel<<<e / 128, 256, 0, stream>>>(H2, Wl1, bl1, Wl3, esrc, edst, out);

  // Reference returns (output, A, weight_norm) -> d_out = [E | N*N | 1] floats.
  hipMemcpyAsync(out + e, A, (size_t)n * (size_t)n * sizeof(float),
                 hipMemcpyDeviceToDevice, stream);
  tail_kernel<<<1, 1, 0, stream>>>(out + (size_t)e + (size_t)n * (size_t)n);
}
